// Indexer_36799279792180
// MI455X (gfx1250) — matrix operand order repaired and checked by reading, not compiled, not run on hardware
//
#include <hip/hip_runtime.h>
#include <hip/hip_bf16.h>
#include <stdint.h>

#define S_LEN  2048
#define D_DIM  2048
#define NHEADS 16
#define HDIM   128
#define TOPK_N 1024

typedef _Float16 h16;
typedef __attribute__((ext_vector_type(16))) _Float16 v16h;
typedef __attribute__((ext_vector_type(8)))  _Float16 v8h;
typedef __attribute__((ext_vector_type(8)))  float    v8f;
typedef __attribute__((ext_vector_type(4)))  unsigned int u32x4;
typedef __attribute__((ext_vector_type(8)))  int         i32x8;
typedef __attribute__((ext_vector_type(4)))  int         i32x4;

// ---- CDNA5 async memory->LDS copy (16B per lane), tracked by ASYNCcnt ----
static __device__ __forceinline__ void async_copy_b128(void* lds_dst, const void* gsrc) {
  uint32_t lds_off = (uint32_t)(uintptr_t)lds_dst;   // ISA: LDS addr = generic addr[31:0]
  uint64_t gaddr   = (uint64_t)(uintptr_t)gsrc;
  asm volatile("global_load_async_to_lds_b128 %0, %1, off"
               :: "v"(lds_off), "v"(gaddr) : "memory");
}
static __device__ __forceinline__ void wait_asynccnt0() {
  asm volatile("s_wait_asynccnt 0" ::: "memory");
}

// ---- CDNA5 Tensor Data Mover: DMA a 2D f16 tile (tile_rows x tile_k) from a
// row-major tensor (row stride = stride_elems) into LDS at pitch tile_k.
// D# per CDNA5 ISA; groups 2/3 zero (2-D tensor).
// amdgpu-toolchain (clang-23) 6-arg builtin form.
static __device__ __forceinline__ void tdm_load_2d_f16(void* lds_dst, const void* gsrc,
                                                       uint32_t tile_k, uint32_t tile_rows,
                                                       uint32_t stride_elems) {
  uint64_t ga = (uint64_t)(uintptr_t)gsrc;
  u32x4 g0;
  g0.x = 1u;                                    // count=1, user descriptor
  g0.y = (uint32_t)(uintptr_t)lds_dst;          // lds_addr = generic addr[31:0]
  g0.z = (uint32_t)ga;                          // global_addr[31:0]
  g0.w = (uint32_t)(ga >> 32) | (2u << 30);     // global_addr[56:32], type=2 (image)
  uint32_t dim0 = stride_elems;                 // tensor_dim0 (elements per row)
  uint32_t dim1 = tile_rows;                    // tensor_dim1 (rows available)
  uint64_t st0  = (uint64_t)stride_elems;       // tensor_dim0_stride
  i32x8 g1;
  g1[0] = (int)(1u << 16);                                      // data_size=1 (2B)
  g1[1] = (int)((dim0 & 0xFFFFu) << 16);                        // tensor_dim0[15:0]
  g1[2] = (int)((dim0 >> 16) | ((dim1 & 0xFFFFu) << 16));       // dim0 hi | dim1 lo
  g1[3] = (int)((dim1 >> 16) | ((tile_k & 0xFFFFu) << 16));     // dim1 hi | tile_dim0
  g1[4] = (int)(tile_rows & 0xFFFFu);                           // tile_dim1 (tile_dim2=0)
  g1[5] = (int)(uint32_t)(st0 & 0xFFFFFFFFu);                   // stride0[31:0]
  g1[6] = (int)(uint32_t)(st0 >> 32);                           // stride0[47:32], stride1=0
  g1[7] = 0;
  i32x4 z4 = {0, 0, 0, 0};
  i32x8 z8 = {0, 0, 0, 0, 0, 0, 0, 0};
  __builtin_amdgcn_tensor_load_to_lds(g0, g1, z4, z4, z8, 0);
}

// ---- WMMA fragment loaders (wave32 layouts from CDNA5 ISA 7.12.2) ----
// A 16x32 f16: lane l: m=l&15, kb=(l<16?0:8); elems 0..7 = K=kb..kb+7, 8..15 = K=kb+16..kb+23
static __device__ __forceinline__ v16h frag_A(const h16* tile, int lane, int pitch) {
  int m  = lane & 15;
  int kb = (lane & 16) ? 8 : 0;
  const h16* p = tile + m * pitch + kb;
  union { v16h v; v8h h[2]; } u;
  u.h[0] = *(const v8h*)(p);
  u.h[1] = *(const v8h*)(p + 16);
  return u.v;
}
// B 32x16 f16 stored in LDS as [n][k]: lane l: n=l&15, kb=(l<16?0:8); elems 0..7 = K=kb..kb+7, 8..15 = K=kb+16..kb+23
static __device__ __forceinline__ v16h frag_B(const h16* tileT, int lane, int pitch) {
  int n  = lane & 15;
  int kb = (lane & 16) ? 8 : 0;
  const h16* p = tileT + n * pitch + kb;
  union { v16h v; v8h h[2]; } u;
  u.h[0] = *(const v8h*)(p);
  u.h[1] = *(const v8h*)(p + 16);
  return u.v;
}

// ---- K0: f32 -> f16 convert ----
__global__ __launch_bounds__(256) void cvt_f16_kernel(const float* __restrict__ src,
                                                      h16* __restrict__ dst, int n) {
  int i = blockIdx.x * 256 + threadIdx.x;
  if (i < n) dst[i] = (h16)src[i];
}

// ---- K0b: f32 [R x K] -> f16 [Rpad x K], zero rows >= R ----
__global__ __launch_bounds__(256) void cvt_pad_f16_kernel(const float* __restrict__ src,
                                                          h16* __restrict__ dst,
                                                          int R, int K, int Rpad) {
  int i = blockIdx.x * 256 + threadIdx.x;
  if (i < Rpad * K) {
    int r = i / K;
    dst[i] = (r < R) ? (h16)src[i] : (h16)0.f;
  }
}

// ---- K2: WMMA GEMM  C[MxN] = A[MxK] * B[NxK]^T  (both row-major f16; B rows = output cols)
// block = 256 threads (8 waves), block tile 128x128, wave tile 32x64, K-step 32.
// Tile staging via Tensor Data Mover (wave 0 issues descriptors; TENSORcnt).
__global__ __launch_bounds__(256) void gemm_f16_kernel(const h16* __restrict__ A,
                                                       const h16* __restrict__ B,
                                                       float* __restrict__ C,
                                                       int N, int K) {
  __shared__ h16 Alds[128 * 32];   // [m][k] pitch 32
  __shared__ h16 Blds[128 * 32];   // [n][k] pitch 32
  const int tid  = threadIdx.x;
  const int lane = tid & 31;
  const int wave = tid >> 5;
  const int m0 = blockIdx.y * 128;
  const int n0 = blockIdx.x * 128;
  const int wm = (wave & 3) * 32;
  const int wn = (wave >> 2) * 64;

  const v8f vzero = {0.f,0.f,0.f,0.f,0.f,0.f,0.f,0.f};
  v8f acc[2][4];
#pragma unroll
  for (int i = 0; i < 2; ++i)
#pragma unroll
    for (int j = 0; j < 4; ++j) acc[i][j] = vzero;

  for (int k0 = 0; k0 < K; k0 += 32) {
    __syncthreads();  // all waves done reading previous tiles
    if (wave == 0) {
      tdm_load_2d_f16(&Alds[0], &A[(size_t)m0 * K + k0], 32, 128, (uint32_t)K);
      tdm_load_2d_f16(&Blds[0], &B[(size_t)n0 * K + k0], 32, 128, (uint32_t)K);
      if (k0 + 32 < K) __builtin_prefetch(&A[(size_t)m0 * K + k0 + 32], 0, 1);
      __builtin_amdgcn_s_wait_tensorcnt(0);
    }
    __syncthreads();  // tiles visible to all waves

    v16h af[2], bfv[4];
    af[0] = frag_A(&Alds[(wm + 0)  * 32], lane, 32);
    af[1] = frag_A(&Alds[(wm + 16) * 32], lane, 32);
#pragma unroll
    for (int j = 0; j < 4; ++j) bfv[j] = frag_B(&Blds[(wn + j * 16) * 32], lane, 32);
#pragma unroll
    for (int ti = 0; ti < 2; ++ti)
#pragma unroll
      for (int tj = 0; tj < 4; ++tj)
        acc[ti][tj] = __builtin_amdgcn_wmma_f32_16x16x32_f16(
            false, af[ti], false, bfv[tj], (short)0, acc[ti][tj], false, false);
  }

  // epilogue (C/D layout: VGPR v -> row v + (lane>=16 ? 8 : 0), col = lane&15)
  const int mb  = (lane & 16) ? 8 : 0;
  const int col = lane & 15;
#pragma unroll
  for (int ti = 0; ti < 2; ++ti)
#pragma unroll
    for (int tj = 0; tj < 4; ++tj) {
      int gcol = n0 + wn + tj * 16 + col;
      if (gcol < N) {
#pragma unroll
        for (int v = 0; v < 8; ++v) {
          int grow = m0 + wm + ti * 16 + mb + v;
          C[(size_t)grow * N + gcol] = acc[ti][tj][v];
        }
      }
    }
}

// ---- K3: RMS-norm + RoPE. grid (S, 17): h<16 -> q head h; h==16 -> k row. block = 128 ----
__global__ __launch_bounds__(128) void norm_rope_kernel(const float* __restrict__ qout,
                                                        const float* __restrict__ kout,
                                                        const float* __restrict__ cosb,
                                                        const float* __restrict__ sinb,
                                                        const float* __restrict__ qnw,
                                                        const float* __restrict__ knw,
                                                        h16* __restrict__ qr,
                                                        h16* __restrict__ kr) {
  __shared__ float red[128];
  const int s = blockIdx.x, h = blockIdx.y, i = threadIdx.x;
  const float* src; const float* nw;
  if (h < NHEADS) { src = qout + (size_t)s * 2048 + h * HDIM; nw = qnw; }
  else            { src = kout + (size_t)s * HDIM;            nw = knw; }
  float v = src[i];
  red[i] = v * v;
  __syncthreads();
  for (int off = 64; off > 0; off >>= 1) {
    if (i < off) red[i] += red[i + off];
    __syncthreads();
  }
  float var = red[0] * (1.0f / 128.0f);
  __syncthreads();
  float xn = v * rsqrtf(var + 1e-6f) * nw[i];
  red[i] = xn;
  __syncthreads();
  float rot = (i < 64) ? -red[i + 64] : red[i - 64];
  float o = xn * cosb[(size_t)s * HDIM + i] + rot * sinb[(size_t)s * HDIM + i];
  if (h < NHEADS) qr[(size_t)s * 2048 + h * HDIM + i] = (h16)o;
  else            kr[(size_t)s * HDIM + i]            = (h16)o;
}

// ---- K3b: scale w by NH^-0.5 * HD^-0.5 ----
__global__ __launch_bounds__(256) void scale_w_kernel(const float* __restrict__ win,
                                                      float* __restrict__ wout, int n) {
  int i = blockIdx.x * 256 + threadIdx.x;
  if (i < n) wout[i] = win[i] * 0.022097086912079612f;
}

// ---- K4: fused score = sum_h relu(Q_h K^T) * w  + causal mask. block tile 128(s) x 64(t)
// Tile staging via per-lane async-to-LDS copies (ASYNCcnt). ----
__global__ __launch_bounds__(256) void score_kernel(const h16* __restrict__ qr,
                                                    const h16* __restrict__ kr,
                                                    const float* __restrict__ wsc,
                                                    float* __restrict__ score) {
  __shared__ h16 Qlds[128 * 128];  // [s][k] for current head (32 KB)
  __shared__ h16 Klds[64 * 128];   // [t][k] (16 KB)
  const int tid  = threadIdx.x;
  const int lane = tid & 31;
  const int wave = tid >> 5;
  const int s0 = blockIdx.y * 128;
  const int t0 = blockIdx.x * 64;
  const int wm = (wave & 3) * 32;
  const int wn = (wave >> 2) * 32;

  // stage K tile once via async copies (1024 chunks of 8 f16)
#pragma unroll
  for (int i = 0; i < 4; ++i) {
    int c  = tid + 256 * i;
    int r  = c >> 4;
    int kk = (c & 15) * 8;
    async_copy_b128(&Klds[r * 128 + kk], &kr[(size_t)(t0 + r) * HDIM + kk]);
  }

  const v8f vzero = {0.f,0.f,0.f,0.f,0.f,0.f,0.f,0.f};
  v8f tot[2][2];
#pragma unroll
  for (int i = 0; i < 2; ++i)
#pragma unroll
    for (int j = 0; j < 2; ++j) tot[i][j] = vzero;

  const int mb = (lane & 16) ? 8 : 0;
  for (int h = 0; h < NHEADS; ++h) {
    __syncthreads();  // previous head done reading Qlds
#pragma unroll
    for (int i = 0; i < 8; ++i) {
      int c  = tid + 256 * i;          // 2048 chunks of 8 f16
      int r  = c >> 4;
      int kk = (c & 15) * 8;
      async_copy_b128(&Qlds[r * 128 + kk], &qr[(size_t)(s0 + r) * 2048 + h * HDIM + kk]);
    }
    wait_asynccnt0();   // covers K-tile copies too on h==0
    __syncthreads();

    v8f acc[2][2];
#pragma unroll
    for (int i = 0; i < 2; ++i)
#pragma unroll
      for (int j = 0; j < 2; ++j) acc[i][j] = vzero;

#pragma unroll
    for (int k0 = 0; k0 < HDIM; k0 += 32) {
      v16h af[2], bfv[2];
      af[0]  = frag_A(&Qlds[(wm + 0)  * 128 + k0], lane, 128);
      af[1]  = frag_A(&Qlds[(wm + 16) * 128 + k0], lane, 128);
      bfv[0] = frag_B(&Klds[(wn + 0)  * 128 + k0], lane, 128);
      bfv[1] = frag_B(&Klds[(wn + 16) * 128 + k0], lane, 128);
#pragma unroll
      for (int ti = 0; ti < 2; ++ti)
#pragma unroll
        for (int tj = 0; tj < 2; ++tj)
          acc[ti][tj] = __builtin_amdgcn_wmma_f32_16x16x32_f16(
              false, af[ti], false, bfv[tj], (short)0, acc[ti][tj], false, false);
    }

    // tot += relu(acc) * w[row][h]
#pragma unroll
    for (int ti = 0; ti < 2; ++ti)
#pragma unroll
      for (int v = 0; v < 8; ++v) {
        int grow = s0 + wm + ti * 16 + mb + v;
        float wv = wsc[(size_t)grow * NHEADS + h];
#pragma unroll
        for (int tj = 0; tj < 2; ++tj) {
          float xv = acc[ti][tj][v];
          tot[ti][tj][v] += fmaxf(xv, 0.0f) * wv;
        }
      }
  }

  // causal mask + store
  const int col = lane & 15;
#pragma unroll
  for (int ti = 0; ti < 2; ++ti)
#pragma unroll
    for (int tj = 0; tj < 2; ++tj)
#pragma unroll
      for (int v = 0; v < 8; ++v) {
        int grow = s0 + wm + ti * 16 + mb + v;
        int gcol = t0 + wn + tj * 16 + col;
        float val = tot[ti][tj][v];
        if (gcol > grow) val = -__builtin_inff();
        score[(size_t)grow * S_LEN + gcol] = val;
      }
}

// ---- K5: per-row top-1024 via full bitonic sort of 2048 kv-pairs in LDS ----
__global__ __launch_bounds__(256) void topk_kernel(const float* __restrict__ score,
                                                   int* __restrict__ out) {
  __shared__ float key[2048];
  __shared__ int   idx[2048];
  const int s = blockIdx.x;
  for (int i = threadIdx.x; i < 2048; i += 256) {
    key[i] = score[(size_t)s * 2048 + i];
    idx[i] = i;
  }
  __syncthreads();
  for (int sz = 2; sz <= 2048; sz <<= 1) {
    for (int j = sz >> 1; j > 0; j >>= 1) {
      for (int i = threadIdx.x; i < 2048; i += 256) {
        int l = i ^ j;
        if (l > i) {
          float ki = key[i], kl = key[l];
          int   ii = idx[i], il = idx[l];
          // strict "precedes" in target order: descending value, ties -> lower index
          bool l_pre_i = (kl > ki) || (kl == ki && il < ii);
          bool i_pre_l = (ki > kl) || (ki == kl && ii < il);
          bool up = ((i & sz) == 0);
          if (up ? l_pre_i : i_pre_l) {
            key[i] = kl; key[l] = ki;
            idx[i] = il; idx[l] = ii;
          }
        }
      }
      __syncthreads();
    }
  }
  for (int i = threadIdx.x; i < TOPK_N; i += 256)
    out[(size_t)s * TOPK_N + i] = idx[i];
}

extern "C" void kernel_launch(void* const* d_in, const int* in_sizes, int n_in,
                              void* d_out, int out_size, void* d_ws, size_t ws_size,
                              hipStream_t stream) {
  const float* x    = (const float*)d_in[0];
  const float* cosb = (const float*)d_in[1];
  const float* sinb = (const float*)d_in[2];
  const float* q_w  = (const float*)d_in[3];
  const float* k_w  = (const float*)d_in[4];
  const float* w_w  = (const float*)d_in[5];
  const float* qnw  = (const float*)d_in[6];
  const float* knw  = (const float*)d_in[7];
  int* out = (int*)d_out;

  char* w = (char*)d_ws;
  const size_t MB = 1024u * 1024u;
  h16*   xb    = (h16*)  (w + 0);                    // 8 MiB   f16 x [s][k]
  h16*   qwf   = (h16*)  (w + 8  * MB);              // 8 MiB   f16 q_w [n][k]
  h16*   kwf   = (h16*)  (w + 16 * MB);              // 512 KiB f16 k_w [n][k]
  h16*   wwf   = (h16*)  (w + 16 * MB + 512 * 1024); // 512 KiB f16 w_w padded to 128 rows
  float* qout  = (float*)(w + 17 * MB);              // 16 MiB
  float* kout  = (float*)(w + 33 * MB);              // 1 MiB
  float* wout  = (float*)(w + 34 * MB);              // 128 KiB
  h16*   qr    = (h16*)  (w + 35 * MB);              // 8 MiB
  h16*   kr    = (h16*)  (w + 43 * MB);              // 512 KiB
  float* wsc   = (float*)(w + 44 * MB);              // 128 KiB
  float* score = (float*)(w + 45 * MB);              // 16 MiB -> 61 MiB total

  const int ND = S_LEN * D_DIM;
  cvt_f16_kernel<<<ND / 256, 256, 0, stream>>>(x, xb, ND);
  cvt_f16_kernel<<<ND / 256, 256, 0, stream>>>(q_w, qwf, ND);
  cvt_f16_kernel<<<(HDIM * D_DIM) / 256, 256, 0, stream>>>(k_w, kwf, HDIM * D_DIM);
  cvt_pad_f16_kernel<<<(128 * D_DIM) / 256, 256, 0, stream>>>(w_w, wwf, NHEADS, D_DIM, 128);

  gemm_f16_kernel<<<dim3(16, 16), 256, 0, stream>>>(xb, qwf, qout, 2048, 2048);
  gemm_f16_kernel<<<dim3(1,  16), 256, 0, stream>>>(xb, kwf, kout, 128,  2048);
  gemm_f16_kernel<<<dim3(1,  16), 256, 0, stream>>>(xb, wwf, wout, 16,   2048);

  norm_rope_kernel<<<dim3(S_LEN, NHEADS + 1), 128, 0, stream>>>(qout, kout, cosb, sinb,
                                                                qnw, knw, qr, kr);
  scale_w_kernel<<<(S_LEN * NHEADS) / 256, 256, 0, stream>>>(wout, wsc, S_LEN * NHEADS);

  score_kernel<<<dim3(S_LEN / 64, S_LEN / 128), 256, 0, stream>>>(qr, kr, wsc, score);
  topk_kernel<<<S_LEN, 256, 0, stream>>>(score, out);
}
